// GraphGymGNN_41317585388128
// MI455X (gfx1250) — matrix operand order repaired and checked by reading, not compiled, not run on hardware
//
#include <hip/hip_runtime.h>
#include <hip/hip_bf16.h>

// ---------------- types for WMMA ----------------
typedef __attribute__((ext_vector_type(16))) _Float16 v16h;
typedef __attribute__((ext_vector_type(8)))  float    v8f;

#define SLOPE 0.2f
#define BN_EPS 1e-5f

// ---- monotone float <-> uint key (so unsigned atomicMax == float max) ----
__device__ __forceinline__ unsigned fkey(float f) {
  unsigned b = __float_as_uint(f);
  return b ^ (unsigned)(((int)b >> 31) | 0x80000000);
}
__device__ __forceinline__ float funkey(unsigned u) {
  unsigned b = (u & 0x80000000u) ? (u ^ 0x80000000u) : ~u;
  return __uint_as_float(b);
}
// key(-inf) = ~0xFF800000 = 0x007FFFFF
#define NEG_INF_KEY 0x007FFFFFu

// =====================================================================
// Weight pack: fp32 W[K=128, NCOLS] -> f16 WMMA B-fragment layout.
// Flat index = ((colTile*(K/32) + kstep)*32 + lane)*16 + e
//   where lane holds col n = colTile*16 + (lane&15), K = kstep*32 + (e>>3)*16 + (lane>>4)*8 + (e&7)
// Each lane's per-kstep fragment becomes 32 contiguous bytes (2x b128 load).
// =====================================================================
template<int NCOLS>
__global__ void pack_w_kernel(const float* __restrict__ W, _Float16* __restrict__ Wp)
{
  int idx = blockIdx.x * blockDim.x + threadIdx.x;
  if (idx >= 128 * NCOLS) return;
  const int e    = idx & 15;
  const int lane = (idx >> 4) & 31;
  const int rest = idx >> 9;
  const int ks   = rest & 3;          // K/32 == 4
  const int ct   = rest >> 2;
  const int n = (ct << 4) + (lane & 15);
  const int k = (ks << 5) + ((e >> 3) << 4) + ((lane >> 4) << 3) + (e & 7);
  Wp[idx] = (_Float16)W[(size_t)k * NCOLS + n];
}

// =====================================================================
// GEMM: C[M,NCOLS] = act( A[M,128] @ W + bias ). One wave32 computes a
// 16x16 tile with 4x v_wmma_f32_16x16x32_f16 (fully unrolled K).
// A loaded as float4 pairs (contiguous K spans), B from packed fragments.
// Block = 128 threads = 4 waves covering 64 output columns.
// =====================================================================
template<int NCOLS>
__global__ __launch_bounds__(128)
void gemm_wmma_kernel(const float* __restrict__ A, const _Float16* __restrict__ Wp,
                      const float* __restrict__ bias, const float* __restrict__ prelu_a,
                      float* __restrict__ C, int M, int act)
{
  constexpr int K = 128;
  const int lane = threadIdx.x & 31;
  const int wave = threadIdx.x >> 5;
  const int row0 = blockIdx.x * 16;
  const int col0 = blockIdx.y * 64 + wave * 16;
  const int colTile = (blockIdx.y << 2) + wave;

  const int m    = lane & 15;
  const int half = lane >> 4;
  const int n    = m;

  int arow = row0 + m; if (arow >= M) arow = M - 1;    // clamp: EXEC stays full
  const float* __restrict__ arowp = A + (size_t)arow * K;
  // lane's packed B fragments: kstep stride = 32 lanes * 16 f16 = 32 v16h
  const v16h* __restrict__ wp =
      (const v16h*)(Wp + ((size_t)(colTile * (K / 32)) * 32 + lane) * 16);

  v8f acc = {};
#pragma unroll
  for (int ks = 0; ks < K / 32; ++ks) {
    v16h af;
#pragma unroll
    for (int g = 0; g < 2; ++g) {
      const float4 a0 = *(const float4*)(arowp + ks * 32 + g * 16 + half * 8);
      const float4 a1 = *(const float4*)(arowp + ks * 32 + g * 16 + half * 8 + 4);
      af[8 * g + 0] = (_Float16)a0.x; af[8 * g + 1] = (_Float16)a0.y;
      af[8 * g + 2] = (_Float16)a0.z; af[8 * g + 3] = (_Float16)a0.w;
      af[8 * g + 4] = (_Float16)a1.x; af[8 * g + 5] = (_Float16)a1.y;
      af[8 * g + 6] = (_Float16)a1.z; af[8 * g + 7] = (_Float16)a1.w;
    }
    const v16h bf = wp[ks * 32];
    acc = __builtin_amdgcn_wmma_f32_16x16x32_f16(false, af, false, bf,
                                                 (short)0, acc, false, false);
  }

  const float pa = prelu_a ? prelu_a[0] : 0.f;
  const float bv = bias[col0 + n];
#pragma unroll
  for (int v = 0; v < 8; ++v) {
    const int mm = row0 + v + (half << 3);   // C layout: lanes 0-15 -> M=v; 16-31 -> M=v+8
    if (mm < M) {
      float val = acc[v] + bv;
      if (act) val = (val >= 0.f) ? val : pa * val;
      C[(size_t)mm * NCOLS + col0 + n] = val;
    }
  }
}

// =====================================================================
// edge_attr mean: sum via wave reduce + atomic
// =====================================================================
__global__ void easum_kernel(const float* __restrict__ eattr, float* __restrict__ easum, int E)
{
  int t = blockIdx.x * blockDim.x + threadIdx.x;
  float v = (t < E) ? eattr[t] : 0.f;
  v += __shfl_xor(v, 1);  v += __shfl_xor(v, 2);  v += __shfl_xor(v, 4);
  v += __shfl_xor(v, 8);  v += __shfl_xor(v, 16);
  if ((threadIdx.x & 31) == 0) atomicAdd(easum, v);
}

// =====================================================================
// Per-layer init: acc = bias (broadcast), ssum = 0, mkey = key(-inf),
// bnsum = 0, easum zeroed only when flagged.
// =====================================================================
__global__ void init_layer_kernel(float* __restrict__ acc, const float* __restrict__ bias,
                                  float* __restrict__ ssum, unsigned* __restrict__ mkey,
                                  float* __restrict__ bnsum, float* __restrict__ easum,
                                  int N, int zero_easum)
{
  int idx = blockIdx.x * blockDim.x + threadIdx.x;
  if (idx < N * 128) acc[idx] = bias[idx & 127];
  if (idx < N * 4) { ssum[idx] = 0.f; mkey[idx] = NEG_INF_KEY; }
  if (idx < 256) bnsum[idx] = 0.f;
  if (idx == 0 && zero_easum) easum[0] = 0.f;
}

// =====================================================================
// GATv2 pass 1: per-edge logits + segment max. One wave32 per edge,
// 4 channels per lane; 8-lane group == one head.
// =====================================================================
__global__ __launch_bounds__(256)
void gat_logits_kernel(const float* __restrict__ xl, const float* __restrict__ xr,
                       const int* __restrict__ ei, const float* __restrict__ eattr,
                       const float* __restrict__ easum, const float* __restrict__ We,
                       const float* __restrict__ att, float* __restrict__ logits,
                       unsigned* __restrict__ mkey, int E, int ET)
{
  const int e    = (blockIdx.x * blockDim.x + threadIdx.x) >> 5;
  const int lane = threadIdx.x & 31;
  if (e >= ET) return;

  int s, d; float ea;
  if (e < E) { s = ei[e]; d = ei[E + e]; ea = eattr[e]; }
  else       { s = e - E; d = s;         ea = easum[0] / (float)E; }

  const int ch = lane << 2;
  float4 vl = *(const float4*)(xl + (size_t)s * 128 + ch);
  float4 vr = *(const float4*)(xr + (size_t)d * 128 + ch);

  float p = 0.f;
#pragma unroll
  for (int i = 0; i < 4; ++i) {
    float z = (&vl.x)[i] + (&vr.x)[i] + ea * We[ch + i];
    z = (z >= 0.f) ? z : SLOPE * z;
    p += z * att[ch + i];                 // att flat over 128 channels (H*C)
  }
  p += __shfl_xor(p, 1);  p += __shfl_xor(p, 2);  p += __shfl_xor(p, 4);

  if ((lane & 7) == 0) {
    const int h = lane >> 3;
    logits[(size_t)e * 4 + h] = p;
    atomicMax(&mkey[(size_t)d * 4 + h], fkey(p));
  }
}

// =====================================================================
// GATv2 pass 2: exp(logit - max) and segment sum. One thread per (e,h).
// =====================================================================
__global__ void gat_exp_kernel(float* __restrict__ logits, const unsigned* __restrict__ mkey,
                               float* __restrict__ ssum, const int* __restrict__ ei,
                               int E, int ET)
{
  int t = blockIdx.x * blockDim.x + threadIdx.x;
  if (t >= ET * 4) return;
  const int e = t >> 2, h = t & 3;
  const int d = (e < E) ? ei[E + e] : (e - E);
  const float mx = funkey(mkey[(size_t)d * 4 + h]);
  const float ex = __expf(logits[t] - mx);
  logits[t] = ex;
  atomicAdd(&ssum[(size_t)d * 4 + h], ex);
}

// =====================================================================
// GATv2 pass 3: acc[d] += alpha * xl[s]. One wave32 per edge.
// =====================================================================
__global__ __launch_bounds__(256)
void gat_scatter_kernel(const float* __restrict__ xl, const float* __restrict__ logits,
                        const float* __restrict__ ssum, const int* __restrict__ ei,
                        float* __restrict__ acc, int E, int ET)
{
  const int e    = (blockIdx.x * blockDim.x + threadIdx.x) >> 5;
  const int lane = threadIdx.x & 31;
  if (e >= ET) return;

  int s, d;
  if (e < E) { s = ei[e]; d = ei[E + e]; }
  else       { s = e - E; d = s; }

  const int h = lane >> 3;
  const float alpha = logits[(size_t)e * 4 + h] / ssum[(size_t)d * 4 + h];

  const int ch = lane << 2;
  float4 vl = *(const float4*)(xl + (size_t)s * 128 + ch);
  float* dst = acc + (size_t)d * 128 + ch;
  atomicAdd(dst + 0, vl.x * alpha);
  atomicAdd(dst + 1, vl.y * alpha);
  atomicAdd(dst + 2, vl.z * alpha);
  atomicAdd(dst + 3, vl.w * alpha);
}

// =====================================================================
// BatchNorm: per-column sum / sumsq (coalesced rows, 128 threads = 128 cols)
// =====================================================================
__global__ __launch_bounds__(128)
void bn_reduce_kernel(const float* __restrict__ h, float* __restrict__ bnsum,
                      int N, int rows_per_block)
{
  const int c  = threadIdx.x;
  int r0 = blockIdx.x * rows_per_block;
  int r1 = r0 + rows_per_block; if (r1 > N) r1 = N;
  float s = 0.f, s2 = 0.f;
  for (int r = r0; r < r1; ++r) {
    float v = h[(size_t)r * 128 + c];
    s += v; s2 += v * v;
  }
  atomicAdd(&bnsum[c], s);
  atomicAdd(&bnsum[128 + c], s2);
}

__global__ void bn_apply_kernel(const float* __restrict__ acc, const float* __restrict__ bnsum,
                                const float* __restrict__ g, const float* __restrict__ beta,
                                const float* __restrict__ prelu_a, float* __restrict__ out, int N)
{
  int idx = blockIdx.x * blockDim.x + threadIdx.x;
  if (idx >= N * 128) return;
  const int c = idx & 127;
  const float inv_n = 1.f / (float)N;
  const float mu  = bnsum[c] * inv_n;
  const float var = bnsum[128 + c] * inv_n - mu * mu;
  float y = (acc[idx] - mu) * __frsqrt_rn(var + BN_EPS) * g[c] + beta[c];
  const float a = prelu_a[0];
  out[idx] = (y >= 0.f) ? y : a * y;
}

// =====================================================================
// host-side launch
// =====================================================================
extern "C" void kernel_launch(void* const* d_in, const int* in_sizes, int n_in,
                              void* d_out, int out_size, void* d_ws, size_t ws_size,
                              hipStream_t stream)
{
  const float* x       = (const float*)d_in[0];
  const int*   ei      = (const int*)  d_in[1];
  const float* eattr   = (const float*)d_in[2];
  const float* pre_W   = (const float*)d_in[3];
  const float* pre_b   = (const float*)d_in[4];
  const float* prelu_a = (const float*)d_in[5];
  const float* post_W  = (const float*)d_in[6];
  const float* post_b  = (const float*)d_in[7];
  const float* out_W   = (const float*)d_in[8];
  const float* out_b   = (const float*)d_in[9];

  const int N  = in_sizes[0] / 128;
  const int E  = in_sizes[2];
  const int ET = E + N;

  // ---- workspace layout (floats; every section is 16B-aligned) ----
  float* ws = (float*)d_ws;
  const size_t nf = (size_t)N * 128;
  float*    b_h    = ws;
  float*    b_xl   = b_h   + nf;
  float*    b_xr   = b_xl  + nf;
  float*    b_acc  = b_xr  + nf;
  float*    b_log  = b_acc + nf;                          // ET*4
  float*    b_ssum = b_log + (size_t)ET * 4;              // N*4
  unsigned* b_mkey = (unsigned*)(b_ssum + (size_t)N * 4); // N*4
  float*    b_bn   = (float*)(b_mkey + (size_t)N * 4);    // 256
  _Float16* b_wp   = (_Float16*)(b_bn + 256);             // 128*128 f16 (8192 floats)
  float*    b_eas  = b_bn + 256 + 8192;                   // 1

  const int gM       = (N + 15) / 16;
  const int eBlocks  = (ET + 7) / 8;
  const int ehBlocks = (ET * 4 + 255) / 256;
  const int nBlocks  = (int)(((size_t)N * 128 + 255) / 256);
  const int bnRows   = 128;
  const int bnBlocks = (N + bnRows - 1) / bnRows;

  // pre-MP: h = prelu(x @ pre_W + pre_b)
  pack_w_kernel<128><<<64, 256, 0, stream>>>(pre_W, b_wp);
  gemm_wmma_kernel<128><<<dim3(gM, 2), 128, 0, stream>>>(x, b_wp, pre_b, prelu_a,
                                                         b_h, N, 1);

  for (int l = 0; l < 2; ++l) {
    const int base = 10 + l * 9;
    const float* Wl   = (const float*)d_in[base + 0];
    const float* bl   = (const float*)d_in[base + 1];
    const float* Wr   = (const float*)d_in[base + 2];
    const float* br   = (const float*)d_in[base + 3];
    const float* We   = (const float*)d_in[base + 4];
    const float* att  = (const float*)d_in[base + 5];
    const float* bias = (const float*)d_in[base + 6];
    const float* g    = (const float*)d_in[base + 7];
    const float* beta = (const float*)d_in[base + 8];

    // node transforms
    pack_w_kernel<128><<<64, 256, 0, stream>>>(Wl, b_wp);
    gemm_wmma_kernel<128><<<dim3(gM, 2), 128, 0, stream>>>(b_h, b_wp, bl, nullptr,
                                                           b_xl, N, 0);
    pack_w_kernel<128><<<64, 256, 0, stream>>>(Wr, b_wp);
    gemm_wmma_kernel<128><<<dim3(gM, 2), 128, 0, stream>>>(b_h, b_wp, br, nullptr,
                                                           b_xr, N, 0);

    // init accumulators (zero edge-attr sum once, first layer)
    init_layer_kernel<<<nBlocks, 256, 0, stream>>>(b_acc, bias, b_ssum, b_mkey,
                                                   b_bn, b_eas, N, l == 0 ? 1 : 0);
    if (l == 0) {
      easum_kernel<<<(E + 255) / 256, 256, 0, stream>>>(eattr, b_eas, E);
    }

    // segment softmax attention + scatter
    gat_logits_kernel<<<eBlocks, 256, 0, stream>>>(b_xl, b_xr, ei, eattr, b_eas,
                                                   We, att, b_log, b_mkey, E, ET);
    gat_exp_kernel<<<ehBlocks, 256, 0, stream>>>(b_log, b_mkey, b_ssum, ei, E, ET);
    gat_scatter_kernel<<<eBlocks, 256, 0, stream>>>(b_xl, b_log, b_ssum, ei,
                                                    b_acc, E, ET);

    // batchnorm + prelu -> next h
    bn_reduce_kernel<<<bnBlocks, 128, 0, stream>>>(b_acc, b_bn, N, bnRows);
    bn_apply_kernel<<<nBlocks, 256, 0, stream>>>(b_acc, b_bn, g, beta, prelu_a,
                                                 b_h, N);
  }

  // post-MP: prelu(h @ post_W + post_b)  (reuse b_xl)
  pack_w_kernel<128><<<64, 256, 0, stream>>>(post_W, b_wp);
  gemm_wmma_kernel<128><<<dim3(gM, 2), 128, 0, stream>>>(b_h, b_wp, post_b, prelu_a,
                                                         b_xl, N, 1);
  // output: b_xl @ out_W + out_b -> d_out  [N,64]
  pack_w_kernel<64><<<32, 256, 0, stream>>>(out_W, b_wp);
  gemm_wmma_kernel<64><<<dim3(gM, 1), 128, 0, stream>>>(b_xl, b_wp, out_b, nullptr,
                                                        (float*)d_out, N, 0);
}
